// SubgraphEmbeddingRegressorModel_62861141344777
// MI455X (gfx1250) — matrix operand order repaired and checked by reading, not compiled, not run on hardware
//
#include <hip/hip_runtime.h>
#include <hip/hip_bf16.h>

// ---------------- types for WMMA ----------------
typedef __attribute__((ext_vector_type(16))) __bf16          v16bf;
typedef __attribute__((ext_vector_type(16))) unsigned short  v16us;
typedef __attribute__((ext_vector_type(8)))  unsigned int    v8u;
typedef __attribute__((ext_vector_type(8)))  float           v8f;

union V16Cast { v16us u; v8u q; v16bf b; };

// round-to-nearest-even f32 -> bf16 bits
static __device__ __forceinline__ unsigned short f32_to_bf16(float f) {
    unsigned int u = __float_as_uint(f);
    unsigned int r = 0x7FFFu + ((u >> 16) & 1u);
    return (unsigned short)((u + r) >> 16);
}

#define C_DIM 256
#define N_NODES 19000
#define N_PAD 19008
#define N_DRUGS 1024

// ---------------- degree: deg[dst[e]] += 1 ----------------
__global__ void degree_kernel(const int* __restrict__ dst, float* __restrict__ deg, int E) {
    int e = blockIdx.x * blockDim.x + threadIdx.x;
    if (e < E) atomicAdd(&deg[dst[e]], 1.0f);
}

// ---------------- dinv = rsqrt(deg + 1) in place ----------------
__global__ void rsqrt_kernel(float* __restrict__ d, int n) {
    int i = blockIdx.x * blockDim.x + threadIdx.x;
    if (i < n) d[i] = rsqrtf(d[i] + 1.0f);
}

// ---------------- pack W[256,256] f32 into WMMA-B bf16 lane layout ----------
// Wpk[((ks*16 + nt)*32 + lane)*16 + e] = bf16( W[(ks*32 + (lane>>4)*16 + e)*256
//                                               + nt*16 + (lane&15)] )
__global__ void pack_w_kernel(const float* __restrict__ W, unsigned short* __restrict__ Wpk) {
    const int i = blockIdx.x * blockDim.x + threadIdx.x;   // 65536 total
    const int e    = i & 15;
    const int lane = (i >> 4) & 31;
    const int nt   = (i >> 9) & 15;
    const int ks   = i >> 13;
    const int k = ks * 32 + (lane >> 4) * 16 + e;
    const int n = nt * 16 + (lane & 15);
    Wpk[i] = f32_to_bf16(W[(size_t)k * C_DIM + n]);
}

// ---------------- pack A[M,256] f32 into WMMA-A bf16 lane layout ------------
// Element offset for (row r, k-step ks, lane-half h, e=0..15):
//   Apk[(r*16 + ks*2 + h)*16 + e] = bf16( A[min(r,valid-1)*256
//                                        + ks*32 + h*16 + e] )
// Each thread packs 8 contiguous elements (one b128 store, two float4 loads).
__global__ void __launch_bounds__(256)
pack_a_kernel(const float* __restrict__ A, unsigned short* __restrict__ Apk, int valid) {
    const int j = blockIdx.x * blockDim.x + threadIdx.x;   // N_PAD*256/8 total
    const int sub  = j & 1;          // (e>>3): which 8-element group
    const int half = (j >> 1) & 1;   // lane half
    const int ks   = (j >> 2) & 7;   // k-step
    const int r    = j >> 5;         // matrix row (padded)
    const int srcr = min(r, valid - 1);
    const int kbase = ks * 32 + half * 16 + sub * 8;
    const float4* ap = (const float4*)(A + (size_t)srcr * C_DIM + kbase);
    float4 f0 = ap[0], f1 = ap[1];
    ushort4 o0, o1;
    o0.x = f32_to_bf16(f0.x); o0.y = f32_to_bf16(f0.y);
    o0.z = f32_to_bf16(f0.z); o0.w = f32_to_bf16(f0.w);
    o1.x = f32_to_bf16(f1.x); o1.y = f32_to_bf16(f1.y);
    o1.z = f32_to_bf16(f1.z); o1.w = f32_to_bf16(f1.w);
    ushort4* op = (ushort4*)(Apk + ((size_t)(r * 16 + ks * 2 + half) * 16 + sub * 8));
    op[0] = o0; op[1] = o1;
}

// ---------------- WMMA GEMM: C[Mpad,256] = Apk @ Wpk (both pre-packed bf16)
// one wave per 64x16 output block (4 accumulator tiles).
__global__ void __launch_bounds__(256)
gemm_wmma_bf16(const unsigned short* __restrict__ Apk,
               const unsigned short* __restrict__ Wpk,
               float* __restrict__ C) {
    const int gwave = (blockIdx.x * blockDim.x + threadIdx.x) >> 5;
    const int lane  = threadIdx.x & 31;
    const int nt = gwave & 15;                 // 16 N-tiles
    const int mb = gwave >> 4;                 // 64-row block index
    const int m0 = mb * 64, n0 = nt * 16;

    const int mA   = lane & 15;
    const int half = lane >> 4;

    // per-lane A base pointers (elements): (r*16 + half)*16 ; +ks*32 per step
    const unsigned short* a0p = Apk + ((size_t)(m0 +  0 + mA) * 16 + half) * 16;
    const unsigned short* a1p = Apk + ((size_t)(m0 + 16 + mA) * 16 + half) * 16;
    const unsigned short* a2p = Apk + ((size_t)(m0 + 32 + mA) * 16 + half) * 16;
    const unsigned short* a3p = Apk + ((size_t)(m0 + 48 + mA) * 16 + half) * 16;
    const unsigned short* bp  = Wpk + ((size_t)nt * 32 + lane) * 16;  // +ks*16*32*16 per step

    v8f acc0 = {}, acc1 = {}, acc2 = {}, acc3 = {};

#pragma unroll
    for (int ks = 0; ks < 8; ++ks) {           // K = 256 in steps of 32
        V16Cast b, a0, a1, a2, a3;
        b.q  = *(const v8u*)(bp  + (size_t)ks * (16 * 32 * 16));
        a0.q = *(const v8u*)(a0p + ks * 32);
        a1.q = *(const v8u*)(a1p + ks * 32);
        a2.q = *(const v8u*)(a2p + ks * 32);
        a3.q = *(const v8u*)(a3p + ks * 32);
        acc0 = __builtin_amdgcn_wmma_f32_16x16x32_bf16(false, a0.b, false, b.b, (short)0, acc0, false, false);
        acc1 = __builtin_amdgcn_wmma_f32_16x16x32_bf16(false, a1.b, false, b.b, (short)0, acc1, false, false);
        acc2 = __builtin_amdgcn_wmma_f32_16x16x32_bf16(false, a2.b, false, b.b, (short)0, acc2, false, false);
        acc3 = __builtin_amdgcn_wmma_f32_16x16x32_bf16(false, a3.b, false, b.b, (short)0, acc3, false, false);
    }

    // D layout: VGPR v -> M = base + v + 8*(lane>>4), N = n0 + (lane&15)
    const int mD = half * 8;
    const int nD = lane & 15;
    float* cp = C + (size_t)(m0 + mD) * C_DIM + n0 + nD;
#pragma unroll
    for (int v = 0; v < 8; ++v) cp[(size_t)(v)      * C_DIM] = acc0[v];
#pragma unroll
    for (int v = 0; v < 8; ++v) cp[(size_t)(v + 16) * C_DIM] = acc1[v];
#pragma unroll
    for (int v = 0; v < 8; ++v) cp[(size_t)(v + 32) * C_DIM] = acc2[v];
#pragma unroll
    for (int v = 0; v < 8; ++v) cp[(size_t)(v + 48) * C_DIM] = acc3[v];
}

// ---------------- edge scatter: agg[dst] += hw[src] * dinv[src]*dinv[dst]
__global__ void __launch_bounds__(256)
spmm_scatter(const int* __restrict__ src, const int* __restrict__ dst,
             const float* __restrict__ dinv, const float* __restrict__ hw,
             float* __restrict__ agg, int E) {
    const long long tid = (long long)blockIdx.x * blockDim.x + threadIdx.x;
    const int e = (int)(tid >> 5);
    const int lane = (int)(tid & 31);
    if (e >= E) return;
    const int s = src[e], d = dst[e];
    const float nrm = dinv[s] * dinv[d];
    const float4* row = (const float4*)(hw + (size_t)s * C_DIM);
    float* out = agg + (size_t)d * C_DIM;
    const int c4 = lane * 2;
    float4 a = row[c4], b = row[c4 + 1];
    const int c = c4 * 4;
    atomicAdd(&out[c + 0], a.x * nrm);
    atomicAdd(&out[c + 1], a.y * nrm);
    atomicAdd(&out[c + 2], a.z * nrm);
    atomicAdd(&out[c + 3], a.w * nrm);
    atomicAdd(&out[c + 4], b.x * nrm);
    atomicAdd(&out[c + 5], b.y * nrm);
    atomicAdd(&out[c + 6], b.z * nrm);
    atomicAdd(&out[c + 7], b.w * nrm);
}

// ---------------- combine: out = agg + hw*dinv^2 + bias [+ resid] [relu]
__global__ void __launch_bounds__(256)
combine_kernel(const float4* __restrict__ agg, const float4* __restrict__ hw,
               const float* __restrict__ dinv, const float4* __restrict__ bias,
               const float4* __restrict__ resid, float4* __restrict__ out,
               int Nvalid, int do_relu) {
    const long long i = (long long)blockIdx.x * blockDim.x + threadIdx.x;
    if (i >= (long long)N_PAD * (C_DIM / 4)) return;
    const int n  = (int)(i >> 6);
    const int c4 = (int)(i & 63);
    if (n >= Nvalid) { out[i] = make_float4(0.f, 0.f, 0.f, 0.f); return; }
    const float sn = dinv[n] * dinv[n];
    float4 a = agg[i], h = hw[i], bb = bias[c4];
    float4 r;
    r.x = a.x + h.x * sn + bb.x;
    r.y = a.y + h.y * sn + bb.y;
    r.z = a.z + h.z * sn + bb.z;
    r.w = a.w + h.w * sn + bb.w;
    if (resid) {
        float4 rr = resid[i];
        r.x += rr.x; r.y += rr.y; r.z += rr.z; r.w += rr.w;
    }
    if (do_relu) {
        r.x = fmaxf(r.x, 0.f); r.y = fmaxf(r.y, 0.f);
        r.z = fmaxf(r.z, 0.f); r.w = fmaxf(r.w, 0.f);
    }
    out[i] = r;
}

// ---------------- pooling counts ----------------
__global__ void count_kernel(const int* __restrict__ gidx, float* __restrict__ cnt, int n) {
    int i = blockIdx.x * blockDim.x + threadIdx.x;
    if (i < n) atomicAdd(&cnt[gidx[i]], 1.0f);
}

// ---------------- pooling scatter: sums[g] += h2[node] ----------------
__global__ void __launch_bounds__(256)
pool_scatter(const int* __restrict__ nodes, const int* __restrict__ gidx,
             const float* __restrict__ h2, float* __restrict__ sums, int n) {
    const long long tid = (long long)blockIdx.x * blockDim.x + threadIdx.x;
    const int i = (int)(tid >> 5);
    const int lane = (int)(tid & 31);
    if (i >= n) return;
    const int nd = nodes[i], g = gidx[i];
    const float4* row = (const float4*)(h2 + (size_t)nd * C_DIM);
    float* out = sums + (size_t)g * C_DIM;
    const int c4 = lane * 2;
    float4 a = row[c4], b = row[c4 + 1];
    const int c = c4 * 4;
    atomicAdd(&out[c + 0], a.x); atomicAdd(&out[c + 1], a.y);
    atomicAdd(&out[c + 2], a.z); atomicAdd(&out[c + 3], a.w);
    atomicAdd(&out[c + 4], b.x); atomicAdd(&out[c + 5], b.y);
    atomicAdd(&out[c + 6], b.z); atomicAdd(&out[c + 7], b.w);
}

// ---------------- mean: emb = sums / max(cnt,1) in place ----------------
__global__ void mean_kernel(float* __restrict__ sums, const float* __restrict__ cnt) {
    const int i = blockIdx.x * blockDim.x + threadIdx.x;
    if (i >= N_DRUGS * C_DIM) return;
    const int g = i >> 8;
    sums[i] = sums[i] / fmaxf(cnt[g], 1.0f);
}

// ---------------- head: out[p] = dot(emb[a[p]], emb[b[p]]) ----------------
__global__ void __launch_bounds__(256)
head_kernel(const int* __restrict__ ia, const int* __restrict__ ib,
            const float* __restrict__ emb, float* __restrict__ out, int B) {
    const long long tid = (long long)blockIdx.x * blockDim.x + threadIdx.x;
    const int p = (int)(tid >> 5);
    const int lane = (int)(tid & 31);
    if (p >= B) return;
    const float4* ea = (const float4*)(emb + (size_t)ia[p] * C_DIM);
    const float4* eb = (const float4*)(emb + (size_t)ib[p] * C_DIM);
    const int c4 = lane * 2;
    float4 a0 = ea[c4], a1 = ea[c4 + 1];
    float4 b0 = eb[c4], b1 = eb[c4 + 1];
    float acc = a0.x * b0.x + a0.y * b0.y + a0.z * b0.z + a0.w * b0.w +
                a1.x * b1.x + a1.y * b1.y + a1.z * b1.z + a1.w * b1.w;
#pragma unroll
    for (int off = 16; off > 0; off >>= 1)
        acc += __shfl_xor(acc, off, 32);
    if (lane == 0) out[p] = acc;
}

extern "C" void kernel_launch(void* const* d_in, const int* in_sizes, int n_in,
                              void* d_out, int out_size, void* d_ws, size_t ws_size,
                              hipStream_t stream) {
    const float* x        = (const float*)d_in[0];
    const int*   ddb      = (const int*)d_in[1];   // [2, B]
    const int*   sg_ei    = (const int*)d_in[4];   // [2, E]
    const int*   sg_nodes = (const int*)d_in[5];
    const int*   sg_avg   = (const int*)d_in[6];
    const float* W1 = (const float*)d_in[7];
    const float* b1 = (const float*)d_in[8];
    const float* W2 = (const float*)d_in[9];
    const float* b2 = (const float*)d_in[10];

    const int E   = in_sizes[4] / 2;
    const int NSG = in_sizes[5];
    const int B   = in_sizes[1] / 2;

    // workspace layout (floats)
    float* ws   = (float*)d_ws;
    float* dinv = ws;                                        // N_PAD
    float* hw   = dinv + N_PAD;                              // N_PAD*256
    float* agg  = hw   + (size_t)N_PAD * C_DIM;              // N_PAD*256
    float* h1   = agg  + (size_t)N_PAD * C_DIM;              // N_PAD*256
    float* sums = h1   + (size_t)N_PAD * C_DIM;              // 1024*256
    float* cnts = sums + (size_t)N_DRUGS * C_DIM;            // 1024
    unsigned short* Wpk1 = (unsigned short*)(cnts + N_DRUGS);            // 65536 u16
    unsigned short* Wpk2 = Wpk1 + (size_t)C_DIM * C_DIM;                 // 65536 u16
    unsigned short* Apk  = Wpk2 + (size_t)C_DIM * C_DIM;                 // N_PAD*256 u16

    const int* src = sg_ei;
    const int* dst = sg_ei + E;

    // degree -> dinv
    hipMemsetAsync(dinv, 0, N_PAD * sizeof(float), stream);
    degree_kernel<<<(E + 255) / 256, 256, 0, stream>>>(dst, dinv, E);
    rsqrt_kernel<<<(N_NODES + 255) / 256, 256, 0, stream>>>(dinv, N_NODES);

    // pre-pack both weight matrices into WMMA-B bf16 layout
    pack_w_kernel<<<(C_DIM * C_DIM) / 256, 256, 0, stream>>>(W1, Wpk1);
    pack_w_kernel<<<(C_DIM * C_DIM) / 256, 256, 0, stream>>>(W2, Wpk2);

    const int gemm_blocks = (N_PAD / 64) * 16 * 32 / 256;    // wave per 64x16 block
    const int packa_blocks = (N_PAD * C_DIM / 8) / 256;
    const long long eThreads = (long long)E * 32;
    const int eBlocks = (int)((eThreads + 255) / 256);
    const long long cThreads = (long long)N_PAD * (C_DIM / 4);
    const int cBlocks = (int)((cThreads + 255) / 256);

    // ---- layer 1 ----
    pack_a_kernel<<<packa_blocks, 256, 0, stream>>>(x, Apk, N_NODES);
    gemm_wmma_bf16<<<gemm_blocks, 256, 0, stream>>>(Apk, Wpk1, hw);
    hipMemsetAsync(agg, 0, (size_t)N_PAD * C_DIM * sizeof(float), stream);
    spmm_scatter<<<eBlocks, 256, 0, stream>>>(src, dst, dinv, hw, agg, E);
    combine_kernel<<<cBlocks, 256, 0, stream>>>(
        (const float4*)agg, (const float4*)hw, dinv, (const float4*)b1,
        nullptr, (float4*)h1, N_NODES, /*relu=*/1);

    // ---- layer 2 (h2 written into agg; residual = h1) ----
    pack_a_kernel<<<packa_blocks, 256, 0, stream>>>(h1, Apk, N_PAD);
    gemm_wmma_bf16<<<gemm_blocks, 256, 0, stream>>>(Apk, Wpk2, hw);
    hipMemsetAsync(agg, 0, (size_t)N_PAD * C_DIM * sizeof(float), stream);
    spmm_scatter<<<eBlocks, 256, 0, stream>>>(src, dst, dinv, hw, agg, E);
    combine_kernel<<<cBlocks, 256, 0, stream>>>(
        (const float4*)agg, (const float4*)hw, dinv, (const float4*)b2,
        (const float4*)h1, (float4*)agg, N_NODES, /*relu=*/0);

    // ---- scatter-mean pooling ----
    hipMemsetAsync(sums, 0, ((size_t)N_DRUGS * C_DIM + N_DRUGS) * sizeof(float), stream);
    count_kernel<<<(NSG + 255) / 256, 256, 0, stream>>>(sg_avg, cnts, NSG);
    const long long pThreads = (long long)NSG * 32;
    pool_scatter<<<(int)((pThreads + 255) / 256), 256, 0, stream>>>(
        sg_nodes, sg_avg, agg, sums, NSG);
    mean_kernel<<<(N_DRUGS * C_DIM + 255) / 256, 256, 0, stream>>>(sums, cnts);

    // ---- dot-product head ----
    const long long hThreads = (long long)B * 32;
    head_kernel<<<(int)((hThreads + 255) / 256), 256, 0, stream>>>(
        ddb, ddb + B, sums, (float*)d_out, B);
}
